// VQSegmentationModel_17480516895134
// MI455X (gfx1250) — compile-verified, matrix operand order checked by reading, not run on hardware
//
#include <hip/hip_runtime.h>
#include <math.h>

// ---- problem constants (from reference) ----
#define B_      32
#define T_      4096
#define HID_    64
#define D_      32
#define KCODES_ 512
#define Q_      4
#define H_      10
#define P_      20
#define N_      4067          // T - H - P + 1
#define DP_     640           // D*P == 2*D*H (fc width)
#define NPAD_   4096          // padded rows per batch for h1

typedef __attribute__((ext_vector_type(16))) _Float16 v16h;
typedef __attribute__((ext_vector_type(8)))  float    v8f;

union Frag { v16h v; unsigned u[8]; };

#define WMMA4(ACC, AF, BF)                                                              \
  do {                                                                                  \
    ACC[0] = __builtin_amdgcn_wmma_f32_16x16x32_f16(false, AF.v, false, BF[0].v,        \
                                                    (short)0, ACC[0], false, false);    \
    ACC[1] = __builtin_amdgcn_wmma_f32_16x16x32_f16(false, AF.v, false, BF[1].v,        \
                                                    (short)0, ACC[1], true,  false);    \
    ACC[2] = __builtin_amdgcn_wmma_f32_16x16x32_f16(false, AF.v, false, BF[2].v,        \
                                                    (short)0, ACC[2], true,  false);    \
    ACC[3] = __builtin_amdgcn_wmma_f32_16x16x32_f16(false, AF.v, false, BF[3].v,        \
                                                    (short)0, ACC[3], true,  false);    \
  } while (0)

__device__ __forceinline__ float wred32(float v) {
#pragma unroll
  for (int m = 16; m >= 1; m >>= 1) v += __shfl_xor(v, m, 32);
  return v;
}

// ---------------- prep: transpose fc weights -> f16 (N,K), zero loss accums ----------------
__global__ void k_prep(const float* __restrict__ f1w, const float* __restrict__ f2w,
                       _Float16* __restrict__ w1T, _Float16* __restrict__ w2T,
                       float* __restrict__ lossbuf) {
  int i = blockIdx.x * 256 + threadIdx.x;
  if (i < 4) lossbuf[i] = 0.0f;
  if (i < DP_ * DP_) {
    int k = i / DP_, n = i % DP_;
    w1T[n * DP_ + k] = (_Float16)f1w[i];
    w2T[n * DP_ + k] = (_Float16)f2w[i];
  }
}

// ---------------- conv1 (2->64, k=7, pad 3) + ReLU; h layout (b,t,c) ----------------
__global__ void k_conv1(const float* __restrict__ traj, const float* __restrict__ w1,
                        const float* __restrict__ b1, float* __restrict__ h) {
  int id = blockIdx.x * 256 + threadIdx.x;   // 0..B*T-1
  int b = id >> 12, t = id & (T_ - 1);
  float x[2][7];
#pragma unroll
  for (int i = 0; i < 2; ++i)
#pragma unroll
    for (int k = 0; k < 7; ++k) {
      int tt = t + k - 3;
      x[i][k] = (tt >= 0 && tt < T_) ? traj[(b * 2 + i) * T_ + tt] : 0.0f;
    }
  float* hp = h + (size_t)id * HID_;
  for (int c = 0; c < HID_; ++c) {
    float acc = b1[c];
    const float* wp = w1 + c * 14;
#pragma unroll
    for (int i = 0; i < 2; ++i)
#pragma unroll
      for (int k = 0; k < 7; ++k) acc += x[i][k] * wp[i * 7 + k];
    hp[c] = fmaxf(acc, 0.0f);
  }
}

// ---------------- conv2 (64->32, k=9, pad 4); z_e layout (b,t,d) ----------------
__global__ __launch_bounds__(256) void k_conv2(const float* __restrict__ h,
                        const float* __restrict__ w2, const float* __restrict__ b2,
                        float* __restrict__ ze) {
  int id = blockIdx.x * 256 + threadIdx.x;
  int b = id >> 12, t = id & (T_ - 1);
  float acc[D_];
#pragma unroll
  for (int c = 0; c < D_; ++c) acc[c] = b2[c];
  for (int k = 0; k < 9; ++k) {
    int tt = t + k - 4;
    if (tt < 0 || tt >= T_) continue;
    const float* hr = h + ((size_t)(b * T_ + tt)) * HID_;
    for (int ci = 0; ci < HID_; ++ci) {
      float v = hr[ci];
      const float* wp = w2 + ci * 9 + k;   // + c*576 strided (wave-uniform -> scalar loads)
#pragma unroll
      for (int c = 0; c < D_; ++c) acc[c] += v * wp[c * 576];
    }
  }
  float* zp = ze + (size_t)id * D_;
#pragma unroll
  for (int c = 0; c < D_; ++c) zp[c] = acc[c];
}

// ---------------- residual VQ w/ rotation trick: one wave32 per (b,t), lane==channel ----------------
__global__ __launch_bounds__(256) void k_quant(const float* __restrict__ ze,
    const float* __restrict__ codebooks, const float* __restrict__ cemb,
    float* __restrict__ quantized, float* __restrict__ tn2, int* __restrict__ c0buf,
    _Float16* __restrict__ zq, float* __restrict__ lossbuf, float* __restrict__ out) {
  int pos  = blockIdx.x * 8 + (threadIdx.x >> 5);   // (b,t) flat, < B*T
  int lane = threadIdx.x & 31;
  float r = ze[(size_t)pos * D_ + lane];
  float quant = 0.0f, vql = 0.0f;
  int c0v = 0;
  for (int qi = 0; qi < Q_; ++qi) {
    const float* cbq = codebooks + (size_t)qi * KCODES_ * D_;
    float rf[32];
#pragma unroll
    for (int j = 0; j < 32; ++j) rf[j] = __shfl(r, j, 32);
    float best = 3.4e38f; int bi = 0;
    for (int cc = lane; cc < KCODES_; cc += 32) {       // 16 codes per lane
      const float* row = cbq + cc * D_;
      float dot = 0.0f, n2 = 0.0f;
#pragma unroll
      for (int j = 0; j < 32; ++j) { float wv = row[j]; dot += rf[j] * wv; n2 += wv * wv; }
      float sc = n2 - 2.0f * dot;                        // ||r||^2 constant -> dropped
      if (sc < best) { best = sc; bi = cc; }
    }
#pragma unroll
    for (int m = 16; m >= 1; m >>= 1) {                  // argmin, first-index tie-break
      float ob = __shfl_xor(best, m, 32);
      int   oi = __shfl_xor(bi,   m, 32);
      if (ob < best || (ob == best && oi < bi)) { best = ob; bi = oi; }
    }
    float qv = cbq[bi * D_ + lane];
    float e  = r;
    float e2 = wred32(e * e);
    float q2 = wred32(qv * qv);
    float u  = e  * rsqrtf(e2 + 1e-12f);
    float qh = qv * rsqrtf(q2 + 1e-12f);
    float s  = u + qh;
    float w  = s * rsqrtf(wred32(s * s) + 1e-12f);
    float ew = wred32(e * w);
    float eu = wred32(e * u);
    float ratio = sqrtf(q2) / (sqrtf(e2) + 1e-12f);
    float rt = ratio * (e - 2.0f * ew * w + 2.0f * eu * qh);
    float df = rt - e;
    vql  += df * df;
    quant += rt;
    r = e - rt;
    if (qi == 0) c0v = bi;
    if (lane == 0) out[(size_t)pos * Q_ + qi] = (float)bi;   // codes output
  }
  quantized[(size_t)pos * D_ + lane] = quant;
  float qn2 = wred32(quant * quant);
  if (lane == 0) { tn2[pos] = qn2; c0buf[pos] = c0v; }
  zq[(size_t)pos * 64 + lane]      = (_Float16)quant;
  zq[(size_t)pos * 64 + 32 + lane] = (_Float16)cemb[c0v * D_ + lane];
  float vsum = wred32(vql);
  if (lane == 0) atomicAdd(&lossbuf[1], vsum);
}

// ---------------- smooth loss ----------------
__global__ void k_smooth(const int* __restrict__ c0, float* __restrict__ lossbuf) {
  int i = blockIdx.x * 256 + threadIdx.x;
  float v = 0.0f;
  if (i < B_ * (T_ - 1)) {
    int b = i / (T_ - 1), t = i % (T_ - 1);
    v = (c0[b * T_ + t] != c0[b * T_ + t + 1]) ? 1.0f : 0.0f;
  }
  v = wred32(v);
  if ((threadIdx.x & 31) == 0) atomicAdd(&lossbuf[2], v);
}

// ======== WMMA GEMM: LDS tile strides (conflict-checked for fragment reads) ========
#define LDA_ 36
#define LDB_ 36

// ---------------- fc1: h1 = relu(z_hist @ W1 + b1), A gathered from overlapping z_q windows ----------------
// Register-prefetch pipeline: global loads for tile kb+1 issue under the WMMAs of tile kb.
__global__ __launch_bounds__(256) void k_fc1(const _Float16* __restrict__ zq,
    const _Float16* __restrict__ w1T, const float* __restrict__ b1,
    _Float16* __restrict__ h1) {
  const int tid = threadIdx.x, lane = tid & 31, wv = tid >> 5;
  const int bidx = blockIdx.y;
  const int n0   = blockIdx.x * 128;
  const int c0t  = blockIdx.z * 64;
  __shared__ _Float16 As[128 * LDA_];
  __shared__ _Float16 Bs[64 * LDB_];
  v8f zero = {0.f,0.f,0.f,0.f,0.f,0.f,0.f,0.f};
  v8f acc[4];
#pragma unroll
  for (int a = 0; a < 4; ++a) acc[a] = zero;

  const int rA = tid >> 1;            // 0..127
  const int kA = (tid & 1) * 16;      // 0 / 16
  const int cB = tid >> 2;            // 0..63
  const int kB = (tid & 3) * 8;       // 0/8/16/24
  const int lane15 = lane & 15;
  const int koff = (lane >> 4) * 8;
  const int rowL = wv * 16 + lane15;
  _Float16* const spA = As + rA * LDA_ + kA;
  _Float16* const spB = Bs + cB * LDB_ + kB;

  // A gather address: A(r,kk) = z_q[b, n0+r+kb/2, (kb&1)*32+kk]  (tail rows clamped)
  auto gaddrA = [&](int kb) {
    int trow = n0 + rA + (kb >> 1);
    if (trow > T_ - 1) trow = T_ - 1;
    return zq + ((size_t)(bidx * T_ + trow)) * 64 + (kb & 1) * 32 + kA;
  };
  uint4 pa0 = *(const uint4*)gaddrA(0);
  uint4 pa1 = *(const uint4*)(gaddrA(0) + 8);
  uint4 pb  = *(const uint4*)(w1T + (size_t)(c0t + cB) * DP_ + kB);

  for (int kb = 0; kb < 20; ++kb) {
    // commit staged registers to LDS
    *(uint2*)(spA)      = make_uint2(pa0.x, pa0.y);
    *(uint2*)(spA + 4)  = make_uint2(pa0.z, pa0.w);
    *(uint2*)(spA + 8)  = make_uint2(pa1.x, pa1.y);
    *(uint2*)(spA + 12) = make_uint2(pa1.z, pa1.w);
    *(uint2*)(spB)      = make_uint2(pb.x, pb.y);
    *(uint2*)(spB + 4)  = make_uint2(pb.z, pb.w);
    __syncthreads();
    // prefetch next tile while WMMAs run
    if (kb < 19) {
      pa0 = *(const uint4*)gaddrA(kb + 1);
      pa1 = *(const uint4*)(gaddrA(kb + 1) + 8);
      pb  = *(const uint4*)(w1T + (size_t)(c0t + cB) * DP_ + (kb + 1) * 32 + kB);
    }
    // fragments: A once, then ALL four B fragments, so the 4 WMMAs issue back-to-back
    Frag Af;
#pragma unroll
    for (int p = 0; p < 8; ++p) {              // 16-bit A layout: K = (i&7) + (i>>3)*16 + (lane>=16)*8
      int K0 = (p & 3) * 2 + (p >> 2) * 16 + koff;
      Af.u[p] = *(const unsigned*)(As + rowL * LDA_ + K0);
    }
    Frag Bf[4];
#pragma unroll
    for (int a = 0; a < 4; ++a) {
      int colL = a * 16 + lane15;
#pragma unroll
      for (int p = 0; p < 8; ++p) {
        int K0 = (p & 3) * 2 + (p >> 2) * 16 + koff;
        Bf[a].u[p] = *(const unsigned*)(Bs + colL * LDB_ + K0);
      }
    }
    WMMA4(acc, Af, Bf);                        // same A operand, reuse_a hint on 2..4
    __syncthreads();
  }
  // epilogue: h1 is padded to 4096 rows/batch -> store unconditionally (no exec churn);
  // rows >= N_ hold finite garbage and are masked in the fc2 loss epilogue.
#pragma unroll
  for (int a = 0; a < 4; ++a) {
    int cg = c0t + a * 16 + lane15;
    float bias = b1[cg];
#pragma unroll
    for (int g = 0; g < 8; ++g) {
      int n = n0 + wv * 16 + g + (lane >> 4) * 8;
      float v = fmaxf(acc[a][g] + bias, 0.0f);
      h1[((size_t)(bidx * NPAD_ + n)) * DP_ + cg] = (_Float16)v;
    }
  }
}

// ---------------- fc2: z_pred tile + fused cosine-sim group loss (z_pred never stored) ----------------
__global__ __launch_bounds__(256) void k_fc2(const _Float16* __restrict__ h1,
    const _Float16* __restrict__ w2T, const float* __restrict__ b2,
    const float* __restrict__ quantized, const float* __restrict__ tn2,
    const int* __restrict__ c0, float* __restrict__ lossbuf) {
  const int tid = threadIdx.x, lane = tid & 31, wv = tid >> 5;
  const int bidx = blockIdx.y;
  const int n0   = blockIdx.x * 128;
  const int c0t  = blockIdx.z * 64;
  __shared__ _Float16 As[128 * LDA_];
  __shared__ _Float16 Bs[64 * LDB_];
  v8f zero = {0.f,0.f,0.f,0.f,0.f,0.f,0.f,0.f};
  v8f acc[4];
#pragma unroll
  for (int a = 0; a < 4; ++a) acc[a] = zero;

  const int rA = tid >> 1;
  const int kA = (tid & 1) * 16;
  const int cB = tid >> 2;
  const int kB = (tid & 3) * 8;
  const int lane15 = lane & 15;
  const int koff = (lane >> 4) * 8;
  const int rowL = wv * 16 + lane15;
  _Float16* const spA = As + rA * LDA_ + kA;
  _Float16* const spB = Bs + cB * LDB_ + kB;
  const _Float16* const gA = h1 + ((size_t)(bidx * NPAD_ + n0 + rA)) * DP_ + kA;
  const _Float16* const gB = w2T + (size_t)(c0t + cB) * DP_ + kB;

  uint4 pa0 = *(const uint4*)gA;
  uint4 pa1 = *(const uint4*)(gA + 8);
  uint4 pb  = *(const uint4*)gB;

  for (int kb = 0; kb < 20; ++kb) {
    *(uint2*)(spA)      = make_uint2(pa0.x, pa0.y);
    *(uint2*)(spA + 4)  = make_uint2(pa0.z, pa0.w);
    *(uint2*)(spA + 8)  = make_uint2(pa1.x, pa1.y);
    *(uint2*)(spA + 12) = make_uint2(pa1.z, pa1.w);
    *(uint2*)(spB)      = make_uint2(pb.x, pb.y);
    *(uint2*)(spB + 4)  = make_uint2(pb.z, pb.w);
    __syncthreads();
    if (kb < 19) {
      pa0 = *(const uint4*)(gA + (kb + 1) * 32);
      pa1 = *(const uint4*)(gA + (kb + 1) * 32 + 8);
      pb  = *(const uint4*)(gB + (kb + 1) * 32);
    }
    Frag Af;
#pragma unroll
    for (int p = 0; p < 8; ++p) {
      int K0 = (p & 3) * 2 + (p >> 2) * 16 + koff;
      Af.u[p] = *(const unsigned*)(As + rowL * LDA_ + K0);
    }
    Frag Bf[4];
#pragma unroll
    for (int a = 0; a < 4; ++a) {
      int colL = a * 16 + lane15;
#pragma unroll
      for (int p = 0; p < 8; ++p) {
        int K0 = (p & 3) * 2 + (p >> 2) * 16 + koff;
        Bf[a].u[p] = *(const unsigned*)(Bs + colL * LDB_ + K0);
      }
    }
    WMMA4(acc, Af, Bf);
    __syncthreads();
  }
  // epilogue: each wave owns rows (16) x two complete p-groups (d=32 each)
  float bias4[4];
#pragma unroll
  for (int a = 0; a < 4; ++a) bias4[a] = b2[c0t + a * 16 + lane15];
  const float* qb = quantized + (size_t)bidx * T_ * D_;
  const float* tb = tn2 + (size_t)bidx * T_;
  const int*   cb = c0  + (size_t)bidx * T_;
  float localLoss = 0.0f;
#pragma unroll
  for (int g = 0; g < 8; ++g) {
    int n = n0 + wv * 16 + g + (lane >> 4) * 8;
    bool valid = (n < N_);
    int  nn = valid ? n : 0;
#pragma unroll
    for (int pg = 0; pg < 2; ++pg) {
      int p = blockIdx.z * 2 + pg;
      int t = H_ + nn + p;
      float pr0 = acc[2 * pg][g]     + bias4[2 * pg];       // d = lane15
      float pr1 = acc[2 * pg + 1][g] + bias4[2 * pg + 1];   // d = 16 + lane15
      float t0 = qb[(size_t)t * D_ + lane15];
      float t1 = qb[(size_t)t * D_ + 16 + lane15];
      float num = pr0 * t0 + pr1 * t1;
      float p2  = pr0 * pr0 + pr1 * pr1;
#pragma unroll
      for (int m = 8; m >= 1; m >>= 1) {                     // 16-lane reduce (rows stay separate)
        num += __shfl_xor(num, m, 32);
        p2  += __shfl_xor(p2,  m, 32);
      }
      if (valid && lane15 == 0) {
        float den = fmaxf(sqrtf(tb[t]) * sqrtf(p2), 1e-8f);
        float sim = num / den;
        float sc  = (cb[H_ + n] == cb[H_ + n + 1]) ? 1.0f : 0.0f;
        float d   = sim - sc;
        localLoss += d * d;
      }
    }
  }
  localLoss = wred32(localLoss);
  if (lane == 0) atomicAdd(&lossbuf[0], localLoss);
}

// ---------------- finalize ----------------
__global__ void k_final(const float* __restrict__ lossbuf, float* __restrict__ out) {
  if (threadIdx.x == 0 && blockIdx.x == 0) {
    float gl = lossbuf[0] / (float)((double)B_ * N_ * P_);
    float vq = lossbuf[1] / (float)((double)Q_ * B_ * T_ * D_);
    float sm = lossbuf[2] / (float)(B_ * (T_ - 1));
    out[(size_t)B_ * T_ * Q_] = gl + vq + sm;
  }
}

extern "C" void kernel_launch(void* const* d_in, const int* in_sizes, int n_in,
                              void* d_out, int out_size, void* d_ws, size_t ws_size,
                              hipStream_t stream) {
  (void)in_sizes; (void)n_in; (void)out_size; (void)ws_size;
  const float* traj = (const float*)d_in[0];
  // d_in[1] masks: unused by the reference forward
  const float* c1w = (const float*)d_in[2];
  const float* c1b = (const float*)d_in[3];
  const float* c2w = (const float*)d_in[4];
  const float* c2b = (const float*)d_in[5];
  const float* f1w = (const float*)d_in[6];
  const float* f1b = (const float*)d_in[7];
  const float* f2w = (const float*)d_in[8];
  const float* f2b = (const float*)d_in[9];
  const float* cemb = (const float*)d_in[10];
  const float* cbks = (const float*)d_in[11];
  float* out = (float*)d_out;

  char* ws = (char*)d_ws;
  size_t off = 0;
  auto alloc = [&](size_t bytes) { void* p = ws + off; off += (bytes + 255) & ~(size_t)255; return p; };
  float*    lossbuf = (float*)alloc(64);
  _Float16* w1T  = (_Float16*)alloc((size_t)DP_ * DP_ * 2);
  _Float16* w2T  = (_Float16*)alloc((size_t)DP_ * DP_ * 2);
  float*    hbuf = (float*)alloc((size_t)B_ * T_ * HID_ * 4);
  float*    ze   = (float*)alloc((size_t)B_ * T_ * D_ * 4);
  float*    quant= (float*)alloc((size_t)B_ * T_ * D_ * 4);
  float*    tn2  = (float*)alloc((size_t)B_ * T_ * 4);
  int*      c0b  = (int*)alloc((size_t)B_ * T_ * 4);
  _Float16* zq   = (_Float16*)alloc((size_t)B_ * T_ * 64 * 2);
  _Float16* h1   = (_Float16*)alloc((size_t)B_ * NPAD_ * DP_ * 2);

  k_prep <<<1600, 256, 0, stream>>>(f1w, f2w, w1T, w2T, lossbuf);
  k_conv1<<<(B_ * T_) / 256, 256, 0, stream>>>(traj, c1w, c1b, hbuf);
  k_conv2<<<(B_ * T_) / 256, 256, 0, stream>>>(hbuf, c2w, c2b, ze);
  k_quant<<<(B_ * T_) / 8, 256, 0, stream>>>(ze, cbks, cemb, quant, tn2, c0b, zq, lossbuf, out);
  k_smooth<<<(B_ * (T_ - 1) + 255) / 256, 256, 0, stream>>>(c0b, lossbuf);
  dim3 gg(32, B_, 10);   // (n-tiles of 128, batch, col-tiles of 64)
  k_fc1<<<gg, 256, 0, stream>>>(zq, w1T, f1b, h1);
  k_fc2<<<gg, 256, 0, stream>>>(h1, w2T, f2b, quant, tn2, c0b, lossbuf);
  k_final<<<1, 64, 0, stream>>>(lossbuf, out);
}
